// MultiHeadAggregation_74217034875167
// MI455X (gfx1250) — matrix operand order repaired and checked by reading, not compiled, not run on hardware
//
#include <hip/hip_runtime.h>
#include <hip/hip_bf16.h>
#include <math.h>

// ---------------------------------------------------------------------------
// MultiHeadAggregation (single-query two-head graph attention), MI455X/gfx1250
//
// Roofline: 2.15 GFLOP vs 67 MB of h traffic -> HBM bound (~3us at 23.3TB/s).
// Strategy: read h once; fold Q into the Wk projection (compat = h @ (NORM*Wk@Q))
// so each 16-row tile needs only 3 bf16 WMMA chains (V_bf, V_af, compat cols);
// online (flash-style) masked softmax per wave; split-merge via workspace.
// ---------------------------------------------------------------------------

#define B_      64
#define N_      2048
#define D_      128
#define KD_     16
#define SPLITS  8
#define ROWS_PER_BLOCK  (N_ / SPLITS)          // 256
#define TILES_PER_BLOCK (ROWS_PER_BLOCK / 16)  // 16
#define NWAVES  4
#define TILES_PER_WAVE  (TILES_PER_BLOCK / NWAVES) // 4
#define NORM_   0.25f                          // 1/sqrt(KEY_DIM=16)
#define NEG_INF (-__builtin_inff())

typedef __attribute__((ext_vector_type(16))) __bf16 v16bf;
typedef __attribute__((ext_vector_type(8)))  float  v8f;

static __device__ __forceinline__ v8f wmma_bf16(v16bf a, v16bf b, v8f c) {
  // D = A(16x32 bf16) x B(32x16 bf16) + C(16x16 f32)
  return __builtin_amdgcn_wmma_f32_16x16x32_bf16(
      /*neg_a=*/false, a, /*neg_b=*/false, b,
      /*c_mod=*/(short)0, c, /*reuse_a=*/false, /*reuse_b=*/false);
}

// ---------------------------------------------------------------------------
// Kernel 1: per (batch, row-split): projections via WMMA + online masked
// softmax; emits per-split partial (mx, sum, head[16]) per attend head.
// ws layout: ws[((b*SPLITS + sp)*2 + att)*18 + {0:mx, 1:sum, 2+n:head[n]}]
// ---------------------------------------------------------------------------
__global__ __launch_bounds__(128)
void mha_attend_kernel(const float* __restrict__ h,
                       const float* __restrict__ adj_c,
                       const int*   __restrict__ phone,
                       const float* __restrict__ Wq_bf, const float* __restrict__ Wk_bf,
                       const float* __restrict__ Wv_bf,
                       const float* __restrict__ Wq_af, const float* __restrict__ Wk_af,
                       const float* __restrict__ Wv_af,
                       float* __restrict__ ws)
{
  __shared__ float s_q[32];                 // Q_bf[16] | Q_af[16]
  __shared__ float s_wqe_bf[D_];            // NORM * (Wk_bf @ Q_bf), folded query
  __shared__ float s_wqe_af[D_];
  __shared__ float s_adj[ROWS_PER_BLOCK];   // adjacency row slice
  __shared__ float r_hp[NWAVES][2][32];     // per-wave head partials
  __shared__ float r_mx[NWAVES][2];
  __shared__ float r_s [NWAVES][2];

  const int b    = blockIdx.y;
  const int sp   = blockIdx.x;
  const int tid  = threadIdx.x;
  const int lane = tid & 31;
  const int wave = tid >> 5;
  const int l16  = lane & 15;
  const int hl   = lane >> 4;               // lane half (K-offset selector)
  const int pset = phone[b];
  const float* hb   = h + (size_t)b * N_ * D_;
  const float* hph  = hb + (size_t)pset * D_;                       // h_phone
  const float* adjr = adj_c + (size_t)b * N_ * N_ + (size_t)pset * N_;
  const int row0 = sp * ROWS_PER_BLOCK;

  // ---- Q = h_phone @ Wq for both heads (tiny: 2 x 128x16 matvec) ----
  if (tid < 32) {
    const float* Wq = (tid < 16) ? Wq_bf : Wq_af;
    const int k = tid & 15;
    float q = 0.f;
    for (int d = 0; d < D_; ++d) q += hph[d] * Wq[d * KD_ + k];
    s_q[tid] = q;
  }
  // ---- stage adjacency slice (coalesced, read once) ----
  for (int i = tid; i < ROWS_PER_BLOCK; i += 128) s_adj[i] = adjr[row0 + i];
  __syncthreads();

  // ---- folded query vectors: wqe = NORM * (Wk @ Q)  (thread d = tid) ----
  {
    float a0 = 0.f, a1 = 0.f;
    for (int k = 0; k < KD_; ++k) {
      a0 += Wk_bf[tid * KD_ + k] * s_q[k];
      a1 += Wk_af[tid * KD_ + k] * s_q[16 + k];
    }
    s_wqe_bf[tid] = a0 * NORM_;
    s_wqe_af[tid] = a1 * NORM_;
  }
  __syncthreads();

  // ---- build persistent B matrices (32x16 chunks, bf16, ISA lane layout:
  //      lane = N (l16), elems 0..7 <- K=8*hl+0..7, elems 8..15 <- K=16+8*hl+0..7) ----
  v16bf Bv_bf[4], Bv_af[4], Bc[4];
#pragma unroll
  for (int c = 0; c < 4; ++c) {
    v16bf b0, b1, b2;
#pragma unroll
    for (int s = 0; s < 16; ++s) {
      const int krow = 32 * c + 8 * hl + (s < 8 ? s : s + 8);
      b0[s] = (__bf16)Wv_bf[krow * KD_ + l16];
      b1[s] = (__bf16)Wv_af[krow * KD_ + l16];
      b2[s] = (__bf16)(l16 == 0 ? s_wqe_bf[krow] : (l16 == 1 ? s_wqe_af[krow] : 0.f));
    }
    Bv_bf[c] = b0; Bv_af[c] = b1; Bc[c] = b2;
  }

  // ---- online softmax state (per wave; mx/sum lane-uniform) ----
  float mx_bf = NEG_INF, sum_bf = 0.f, hp_bf = 0.f;   // hp: n=l16, m in hl*8..+7
  float mx_af = NEG_INF, sum_af = 0.f, hp_af = 0.f;

  for (int tt = 0; tt < TILES_PER_WAVE; ++tt) {
    const int t = wave + NWAVES * tt;           // tile index within block
    const int rbase = row0 + t * 16;
    const float* arow = hb + (size_t)(rbase + l16) * D_;

    // prefetch next tile's row for this lane (global_prefetch_b8)
    {
      const int nr = rbase + NWAVES * 16 + l16;
      if (nr < N_) __builtin_prefetch(hb + (size_t)nr * D_, 0, 0);
    }

    v8f accV_bf = {0,0,0,0,0,0,0,0};
    v8f accV_af = {0,0,0,0,0,0,0,0};
    v8f accC    = {0,0,0,0,0,0,0,0};
#pragma unroll
    for (int c = 0; c < 4; ++c) {
      // A tile 16x32 bf16: lane l16 = row; elems 0..7 <- K=32c+8*hl+0..7,
      // elems 8..15 <- K=32c+16+8*hl+0..7 (ISA 7.12.2 16-bit A layout)
      const float4* p = (const float4*)(arow + 32 * c + 8 * hl);
      const float4 x0 = p[0], x1 = p[1];        // +0 .. +7
      const float4 x2 = p[4], x3 = p[5];        // +16 .. +23
      v16bf A;
      A[0]=(__bf16)x0.x;  A[1]=(__bf16)x0.y;  A[2]=(__bf16)x0.z;  A[3]=(__bf16)x0.w;
      A[4]=(__bf16)x1.x;  A[5]=(__bf16)x1.y;  A[6]=(__bf16)x1.z;  A[7]=(__bf16)x1.w;
      A[8]=(__bf16)x2.x;  A[9]=(__bf16)x2.y;  A[10]=(__bf16)x2.z; A[11]=(__bf16)x2.w;
      A[12]=(__bf16)x3.x; A[13]=(__bf16)x3.y; A[14]=(__bf16)x3.z; A[15]=(__bf16)x3.w;
      accV_bf = wmma_bf16(A, Bv_bf[c], accV_bf);
      accV_af = wmma_bf16(A, Bv_af[c], accV_af);
      accC    = wmma_bf16(A, Bc[c],    accC);
    }

    // ---- extract compat (C layout: elem r, lane L -> M=r+8*(L/16), N=L%16),
    //      apply 10*tanh clip + mask at owner lanes (N=0: bf, N=1: af) ----
    float myc[8];
#pragma unroll
    for (int r = 0; r < 8; ++r) myc[r] = NEG_INF;
    if (l16 < 2) {
#pragma unroll
      for (int r = 0; r < 8; ++r) {
        const int m   = r + 8 * hl;
        const int pos = rbase + m;
        const float adjv = s_adj[t * 16 + m];
        const bool inm = (l16 == 0) ? (pos < pset) : (pos >= pset);
        const float cv = 10.f * tanhf(accC[r]);
        myc[r] = (inm && adjv != 0.f) ? cv : NEG_INF;
      }
    }
    // broadcast the 16 compat values of each head to all lanes
    float cbf[16], caf[16];
#pragma unroll
    for (int m = 0; m < 16; ++m) {
      const int sl = (m >> 3) * 16;
      cbf[m] = __shfl(myc[m & 7], sl,     32);
      caf[m] = __shfl(myc[m & 7], sl + 1, 32);
    }

    // ---- online update, bf head ----
    {
      float tmx = NEG_INF;
#pragma unroll
      for (int m = 0; m < 16; ++m) tmx = fmaxf(tmx, cbf[m]);
      if (tmx > NEG_INF) {
        const float nmx = fmaxf(mx_bf, tmx);
        const float sc  = __expf(mx_bf - nmx);       // exp(-inf)=0 on first hit
        float e[16]; float se = 0.f;
#pragma unroll
        for (int m = 0; m < 16; ++m) { e[m] = __expf(cbf[m] - nmx); se += e[m]; }
        float hv = 0.f;
#pragma unroll
        for (int r = 0; r < 8; ++r) {
          const int m = r + 8 * hl;
          hv += e[m] * s_adj[t * 16 + m] * accV_bf[r];  // attn * adj fidelity
        }
        sum_bf = sum_bf * sc + se;
        hp_bf  = hp_bf  * sc + hv;
        mx_bf  = nmx;
      }
    }
    // ---- online update, af head ----
    {
      float tmx = NEG_INF;
#pragma unroll
      for (int m = 0; m < 16; ++m) tmx = fmaxf(tmx, caf[m]);
      if (tmx > NEG_INF) {
        const float nmx = fmaxf(mx_af, tmx);
        const float sc  = __expf(mx_af - nmx);
        float e[16]; float se = 0.f;
#pragma unroll
        for (int m = 0; m < 16; ++m) { e[m] = __expf(caf[m] - nmx); se += e[m]; }
        float hv = 0.f;
#pragma unroll
        for (int r = 0; r < 8; ++r) {
          const int m = r + 8 * hl;
          hv += e[m] * s_adj[t * 16 + m] * accV_af[r];
        }
        sum_af = sum_af * sc + se;
        hp_af  = hp_af  * sc + hv;
        mx_af  = nmx;
      }
    }
  }

  // ---- merge the 4 waves of this block, emit one partial per attend ----
  r_hp[wave][0][lane] = hp_bf;
  r_hp[wave][1][lane] = hp_af;
  if (lane == 0) {
    r_mx[wave][0] = mx_bf; r_s[wave][0] = sum_bf;
    r_mx[wave][1] = mx_af; r_s[wave][1] = sum_af;
  }
  __syncthreads();
  if (tid < 32) {
    const int att = tid >> 4, n = tid & 15;
    float MX = NEG_INF;
    for (int w = 0; w < NWAVES; ++w) MX = fmaxf(MX, r_mx[w][att]);
    float S = 0.f, H = 0.f;
    if (MX > NEG_INF) {
      for (int w = 0; w < NWAVES; ++w) {
        const float m = r_mx[w][att];
        if (m > NEG_INF) {
          const float e = __expf(m - MX);
          S += r_s[w][att] * e;
          H += (r_hp[w][att][n] + r_hp[w][att][n + 16]) * e;
        }
      }
    }
    float* slot = ws + (((size_t)b * SPLITS + sp) * 2 + att) * 18;
    if (n == 0) { slot[0] = MX; slot[1] = S; }
    slot[2 + n] = H;
  }
}

// ---------------------------------------------------------------------------
// Kernel 2: merge the SPLITS partials per (batch, attend), normalize, and
// apply the 16 -> 128 output projection. d_out = [g_bf (64x128), g_af (64x128)]
// ---------------------------------------------------------------------------
__global__ __launch_bounds__(128)
void mha_finalize_kernel(const float* __restrict__ ws,
                         const float* __restrict__ Wo_bf,
                         const float* __restrict__ Wo_af,
                         float* __restrict__ out)
{
  __shared__ float sh_head[32];
  const int b = blockIdx.x;
  const int tid = threadIdx.x;

  if (tid < 32) {
    const int att = tid >> 4, n = tid & 15;
    float MX = NEG_INF;
    for (int sp = 0; sp < SPLITS; ++sp)
      MX = fmaxf(MX, ws[(((size_t)b * SPLITS + sp) * 2 + att) * 18]);
    float S = 0.f, H = 0.f;
    if (MX > NEG_INF) {
      for (int sp = 0; sp < SPLITS; ++sp) {
        const float* sl = ws + (((size_t)b * SPLITS + sp) * 2 + att) * 18;
        const float m = sl[0];
        if (m > NEG_INF) {
          const float e = __expf(m - MX);
          S += sl[1] * e;
          H += sl[2 + n] * e;
        }
      }
    }
    sh_head[tid] = (S > 0.f) ? (H / S) : 0.f;   // fully-masked rows -> zeros
  }
  __syncthreads();

  float o0 = 0.f, o1 = 0.f;
#pragma unroll
  for (int n = 0; n < KD_; ++n) {
    o0 += sh_head[n]      * Wo_bf[n * D_ + tid];
    o1 += sh_head[16 + n] * Wo_af[n * D_ + tid];
  }
  out[(size_t)b * D_ + tid]                    = o0;
  out[(size_t)B_ * D_ + (size_t)b * D_ + tid]  = o1;
}

// ---------------------------------------------------------------------------
extern "C" void kernel_launch(void* const* d_in, const int* in_sizes, int n_in,
                              void* d_out, int out_size, void* d_ws, size_t ws_size,
                              hipStream_t stream) {
  (void)in_sizes; (void)n_in; (void)out_size; (void)ws_size;
  const float* h     = (const float*)d_in[0];
  const float* adj_c = (const float*)d_in[1];
  const int*   phone = (const int*)  d_in[2];
  const float* Wq_bf = (const float*)d_in[3];
  const float* Wk_bf = (const float*)d_in[4];
  const float* Wv_bf = (const float*)d_in[5];
  const float* Wo_bf = (const float*)d_in[6];
  const float* Wq_af = (const float*)d_in[7];
  const float* Wk_af = (const float*)d_in[8];
  const float* Wv_af = (const float*)d_in[9];
  const float* Wo_af = (const float*)d_in[10];
  float* ws  = (float*)d_ws;   // needs 64*8*2*18 floats = 73,728 B
  float* out = (float*)d_out;

  dim3 g1(SPLITS, B_);
  mha_attend_kernel<<<g1, 128, 0, stream>>>(h, adj_c, phone,
                                            Wq_bf, Wk_bf, Wv_bf,
                                            Wq_af, Wk_af, Wv_af, ws);
  mha_finalize_kernel<<<B_, 128, 0, stream>>>(ws, Wo_bf, Wo_af, out);
}
